// MultiHeadDenseSynthesizer_87488483820311
// MI455X (gfx1250) — matrix operand order repaired and checked by reading, not compiled, not run on hardware
//
#include <hip/hip_runtime.h>

// ---------------------------------------------------------------------------
// MultiHeadDenseSynthesizer for MI455X (gfx1250, wave32, WMMA bf16)
// All B operands pre-transposed to bf16 column-major so fragment loads are
// contiguous 32B per lane (two b128 loads). No scalar ds_load_u16 paths.
// ---------------------------------------------------------------------------

typedef __attribute__((ext_vector_type(16))) __bf16        v16bf;
typedef __attribute__((ext_vector_type(8)))  float         v8f;
typedef __attribute__((ext_vector_type(4)))  unsigned int  u32x4;

#define NB 64      // batch
#define LL 500     // seq len
#define LP 512     // padded seq len
#define FF 256     // features
#define HH 4       // heads
#define DK 64      // head dim

// ---- helpers --------------------------------------------------------------

__device__ __forceinline__ unsigned short f2bf_u(float f) {
  return __builtin_bit_cast(unsigned short, (__bf16)f);
}

struct U8x { u32x4 a, b; };

__device__ __forceinline__ v16bf frag_from(const unsigned short* p0,
                                           const unsigned short* p1) {
  U8x u;
  u.a = *(const u32x4*)p0;   // elements 0..7
  u.b = *(const u32x4*)p1;   // elements 8..15
  return __builtin_bit_cast(v16bf, u);
}

// A fragment: 16x32 bf16 tile, row-major source (stride in elements, mult of 8).
// lanes 0-15 row=lane, K=0..15; lanes 16-31 K=16..31 (matches B frag order).
__device__ __forceinline__ v16bf load_a_frag(const unsigned short* base, int stride) {
  int lane = threadIdx.x & 31;
  const unsigned short* p = base + (lane & 15) * stride + ((lane >> 4) << 4);
  return frag_from(p, p + 8);
}

// B fragment from column-major [N][K] source: lane c covers col c, 16
// contiguous K values starting at kb (0 or 16).  ldk must be multiple of 16.
__device__ __forceinline__ v16bf load_b_frag_cm(const unsigned short* base, int ldk) {
  int lane = threadIdx.x & 31;
  const unsigned short* p = base + (lane & 15) * ldk + ((lane >> 4) << 4);
  return frag_from(p, p + 8);
}

__device__ __forceinline__ v8f wmma_bf16(v16bf a, v16bf b, v8f c) {
  return __builtin_amdgcn_wmma_f32_16x16x32_bf16(false, a, false, b,
                                                 (short)0, c, false, false);
}

__device__ __forceinline__ float red8_sum(float x) {
  x += __shfl_xor(x, 1, 32);
  x += __shfl_xor(x, 2, 32);
  x += __shfl_xor(x, 4, 32);
  return x;
}
__device__ __forceinline__ float red8_max(float x) {
  x = fmaxf(x, __shfl_xor(x, 1, 32));
  x = fmaxf(x, __shfl_xor(x, 2, 32));
  x = fmaxf(x, __shfl_xor(x, 4, 32));
  return x;
}

// ---- kernel: bf16 transpose of a weight  dst[n*K+k] = src[k*N+n] ----------
// (zero-fills n in [N, Nout) so padded B tiles need no bound checks)

__global__ __launch_bounds__(256) void transpose_bf16_kernel(
    const float* __restrict__ src, unsigned short* __restrict__ dst,
    int K, int N, int Nout) {
  int idx = blockIdx.x * 256 + threadIdx.x;
  if (idx >= Nout * K) return;
  int n = idx / K, k = idx - n * K;
  dst[idx] = (n < N) ? f2bf_u(src[(size_t)k * N + n]) : (unsigned short)0;
}

// ---- kernel: zero pad rows of qh (row-major) and vh_t (col-major) --------

__global__ __launch_bounds__(256) void pad_zero_kernel(unsigned short* __restrict__ qh,
                                                       unsigned short* __restrict__ vh_t) {
  int bh = blockIdx.x;                               // 0..255
  // qh: rows LL..LP-1 contiguous
  size_t qbase = (size_t)bh * LP * DK + (size_t)LL * DK;
  for (int i = threadIdx.x; i < (LP - LL) * DK; i += 256) qh[qbase + i] = 0;
  // vh_t: per column d, rows LL..LP-1 (stride LP)
  size_t vbase = (size_t)bh * DK * LP;
  for (int i = threadIdx.x; i < DK * (LP - LL); i += 256) {
    int d = i / (LP - LL), l = LL + i % (LP - LL);
    vh_t[vbase + (size_t)d * LP + l] = 0;
  }
}

// ---- kernel 1: projection GEMM  (X @ W) -> head layout --------------------
// X: [B*L, F] f32;  Wt: bf16 col-major [F][F].  Block tile 64x64, 4 waves.
// colmajor_out==0: out[b][h][l][d] (A-operand layout for qh)
// colmajor_out==1: out[b][h][d][l] (B-operand layout for vh)

__global__ __launch_bounds__(128) void proj_kernel(const float* __restrict__ X,
                                                   const unsigned short* __restrict__ Wt,
                                                   unsigned short* __restrict__ out,
                                                   int colmajor_out) {
  __shared__ unsigned short As[64 * 32];             // 4 KB
  const int m0  = blockIdx.x * 64;
  const int n0  = blockIdx.y * 64;
  const int tid = threadIdx.x;
  const int w   = tid >> 5;

  v8f acc[4] = {};
  for (int k0 = 0; k0 < FF; k0 += 32) {
    // stage A 64x32 (f32 -> bf16), vectorized: float4 load, uint2 LDS store
#pragma unroll
    for (int i = 0; i < 4; ++i) {
      int e  = tid + i * 128;                        // 0..511 float4 groups
      int r  = e >> 3, c4 = (e & 7) * 4;
      float4 f = *(const float4*)(X + (size_t)(m0 + r) * FF + k0 + c4);
      uint2 pk;
      pk.x = (unsigned)f2bf_u(f.x) | ((unsigned)f2bf_u(f.y) << 16);
      pk.y = (unsigned)f2bf_u(f.z) | ((unsigned)f2bf_u(f.w) << 16);
      *(uint2*)(As + r * 32 + c4) = pk;
    }
    __syncthreads();
    v16bf a = load_a_frag(As + (w * 16) * 32, 32);
#pragma unroll
    for (int j = 0; j < 4; ++j) {
      v16bf b = load_b_frag_cm(Wt + (size_t)(n0 + j * 16) * FF + k0, FF);
      acc[j] = wmma_bf16(a, b, acc[j]);
    }
    __syncthreads();
  }

  int lane = tid & 31, c = lane & 15, rb = (lane >> 4) << 3;
#pragma unroll
  for (int j = 0; j < 4; ++j) {
    int gn = n0 + j * 16 + c;
    int h = gn >> 6, d = gn & 63;
#pragma unroll
    for (int jj = 0; jj < 8; ++jj) {
      int gm = m0 + w * 16 + rb + jj;                // < 32000 always
      int bb = gm / LL, l = gm % LL;
      size_t off = colmajor_out
                       ? (((size_t)bb * HH + h) * DK + d) * LP + l
                       : (((size_t)bb * HH + h) * LP + l) * DK + d;
      out[off] = f2bf_u(acc[j][jj]);
    }
  }
}

// ---- kernel 2: fused synthesizer + softmax + attn@vh ----------------------
// grid (32 L-tiles, 256 bh); 128 threads = 4 waves.

__global__ __launch_bounds__(128) void synth_kernel(const unsigned short* __restrict__ qh,
                                                    const unsigned short* __restrict__ vh_t,
                                                    const unsigned short* __restrict__ w1_t,
                                                    const float* __restrict__ b1,
                                                    const unsigned short* __restrict__ w2_t,
                                                    const float* __restrict__ b2,
                                                    unsigned short* __restrict__ oh) {
  __shared__ float          Lg[16 * LP];             // 32 KB logits/probs f32
  __shared__ unsigned short Pb[16 * LP];             // 16 KB probs bf16
  __shared__ unsigned short Wb[16 * DK];             //  2 KB relu(qh@w1+b1)

  const int bh   = blockIdx.y;
  const int l0   = blockIdx.x * 16;
  const int tid  = threadIdx.x;
  const int w    = tid >> 5;
  const int lane = tid & 31;
  const int c    = lane & 15;
  const int rb   = (lane >> 4) << 3;
  const size_t qbase = (size_t)bh * LP * DK;         // qh row-major base
  const size_t vbase = (size_t)bh * DK * LP;         // vh_t col-major base

  // --- stage 1: W = relu(qh_tile @ w1 + b1); wave w -> cols 16w..16w+15 ----
  v8f accw = {};
#pragma unroll
  for (int ks = 0; ks < 2; ++ks) {
    v16bf a = load_a_frag(qh + qbase + (size_t)l0 * DK + ks * 32, DK);
    v16bf b = load_b_frag_cm(w1_t + (w * 16) * DK + ks * 32, DK);
    accw = wmma_bf16(a, b, accw);
  }
  {
    float bias = b1[w * 16 + c];
#pragma unroll
    for (int j = 0; j < 8; ++j) {
      float x = accw[j] + bias;
      Wb[(rb + j) * DK + w * 16 + c] = f2bf_u(x > 0.f ? x : 0.f);
    }
  }
  __syncthreads();

  // --- stage 2: logits[16,512] = W @ w2 + b2 (pad cols of w2_t are zero) ---
  v8f accl[8] = {};
#pragma unroll
  for (int ks = 0; ks < 2; ++ks) {
    v16bf a = load_a_frag(Wb + ks * 32, DK);
#pragma unroll
    for (int j = 0; j < 8; ++j) {
      int nt = w * 8 + j;
      v16bf b = load_b_frag_cm(w2_t + (size_t)(nt * 16) * DK + ks * 32, DK);
      accl[j] = wmma_bf16(a, b, accl[j]);
    }
  }
#pragma unroll
  for (int j = 0; j < 8; ++j) {
    int gn = (w * 8 + j) * 16 + c;
    float bias = (gn < LL) ? b2[gn] : 0.f;
#pragma unroll
    for (int jj = 0; jj < 8; ++jj) {
      float x = accl[j][jj] + bias;
      Lg[(rb + jj) * LP + gn] = (gn < LL) ? x : -1e30f;
    }
  }
  // prefetch this block's vh_t panel while softmax runs
  __builtin_prefetch((const void*)(vh_t + vbase + (size_t)tid * 256), 0, 1);
  __builtin_prefetch((const void*)(vh_t + vbase + 32768 + (size_t)tid * 256), 0, 1);
  __syncthreads();

  // --- softmax over 512 cols, 8 threads per row ----------------------------
  {
    int row = tid >> 3, sub = tid & 7;
    float* R = Lg + row * LP + sub * 64;
    float m = -1e30f;
#pragma unroll
    for (int i = 0; i < 64; ++i) m = fmaxf(m, R[i]);
    m = red8_max(m);
    float s = 0.f;
#pragma unroll
    for (int i = 0; i < 64; ++i) {
      float p = __expf(R[i] - m);
      s += p;
      R[i] = p;
    }
    s = red8_sum(s);
    float inv = 1.f / s;
    unsigned int* P32 = (unsigned int*)(Pb + row * LP + sub * 64);
#pragma unroll
    for (int i = 0; i < 32; ++i)
      P32[i] = (unsigned)f2bf_u(R[2 * i] * inv) |
               ((unsigned)f2bf_u(R[2 * i + 1] * inv) << 16);
  }
  __syncthreads();

  // --- stage 3: O[16,64] = P @ vh; B frags direct from col-major vh_t ------
  v8f acco = {};
  const unsigned short* vcol = vh_t + vbase + (size_t)(w * 16) * LP;
#pragma unroll
  for (int k0 = 0; k0 < LP; k0 += 32) {
    v16bf a = load_a_frag(Pb + k0, LP);
    v16bf b = load_b_frag_cm(vcol + k0, LP);
    acco = wmma_bf16(a, b, acco);
  }

  // write bf16 in [B, L, H, DK] (row-major [B*L, F]) layout
  int bb = bh >> 2, h = bh & 3;
#pragma unroll
  for (int j = 0; j < 8; ++j) {
    int l = l0 + rb + j;
    if (l < LL)
      oh[(((size_t)bb * LL + l) * HH + h) * DK + w * 16 + c] = f2bf_u(acco[j]);
  }
}

// ---- kernel 3: fc GEMM + residual + LayerNorm -----------------------------
// 16 rows x 256 cols per block; 4 waves, wave w -> cols 64w..64w+63.

__global__ __launch_bounds__(128) void fc_ln_kernel(const unsigned short* __restrict__ oh,
                                                    const unsigned short* __restrict__ fcw_t,
                                                    const float* __restrict__ q,
                                                    const float* __restrict__ g,
                                                    const float* __restrict__ beta,
                                                    float* __restrict__ out) {
  __shared__ float Os[16 * FF];                      // 16 KB
  const int m0   = blockIdx.x * 16;
  const int tid  = threadIdx.x;
  const int w    = tid >> 5;
  const int lane = tid & 31;
  const int c    = lane & 15;
  const int rb   = (lane >> 4) << 3;

  v8f acc[4] = {};
#pragma unroll
  for (int k0 = 0; k0 < FF; k0 += 32) {
    v16bf a = load_a_frag(oh + (size_t)m0 * FF + k0, FF);
#pragma unroll
    for (int j = 0; j < 4; ++j) {
      v16bf b = load_b_frag_cm(fcw_t + (size_t)(w * 64 + j * 16) * FF + k0, FF);
      acc[j] = wmma_bf16(a, b, acc[j]);
    }
  }
#pragma unroll
  for (int j = 0; j < 4; ++j) {
    int gn = w * 64 + j * 16 + c;
#pragma unroll
    for (int jj = 0; jj < 8; ++jj)
      Os[(rb + jj) * FF + gn] = acc[j][jj];
  }
  __syncthreads();

  // LayerNorm: 8 threads per row, 32 contiguous cols each; residual added here
  int row = tid >> 3, sub = tid & 7;
  int gm = m0 + row;
  const float* R = Os + row * FF + sub * 32;
  const float* Q = q + (size_t)gm * FF + sub * 32;
  float xs[32];
  float s1 = 0.f, s2 = 0.f;
#pragma unroll
  for (int i = 0; i < 32; ++i) {
    float x = R[i] + Q[i];
    xs[i] = x;
    s1 += x;
    s2 += x * x;
  }
  s1 = red8_sum(s1);
  s2 = red8_sum(s2);
  float mu  = s1 * (1.f / FF);
  float var = s2 * (1.f / FF) - mu * mu;
  float rs  = rsqrtf(var + 1e-6f);
  float* O = out + (size_t)gm * FF + sub * 32;
  const float* G = g + sub * 32;
  const float* Bt = beta + sub * 32;
#pragma unroll
  for (int i = 0; i < 32; ++i)
    O[i] = (xs[i] - mu) * rs * G[i] + Bt[i];
}

// ---- launch ---------------------------------------------------------------

extern "C" void kernel_launch(void* const* d_in, const int* in_sizes, int n_in,
                              void* d_out, int out_size, void* d_ws, size_t ws_size,
                              hipStream_t stream) {
  const float* q    = (const float*)d_in[0];
  // d_in[1] (k) is unused by the reference
  const float* v    = (const float*)d_in[2];
  const float* w_qs = (const float*)d_in[3];
  const float* w_vs = (const float*)d_in[4];
  const float* w1   = (const float*)d_in[5];
  const float* b1   = (const float*)d_in[6];
  const float* w2   = (const float*)d_in[7];
  const float* b2   = (const float*)d_in[8];
  const float* fc_w = (const float*)d_in[9];
  const float* ln_g = (const float*)d_in[10];
  const float* ln_b = (const float*)d_in[11];

  const size_t headElems = (size_t)NB * HH * LP * DK;     // 8,388,608
  unsigned short* qh    = (unsigned short*)d_ws;
  unsigned short* vh_t  = qh + headElems;
  unsigned short* oh    = vh_t + headElems;               // [B*L, F] bf16
  unsigned short* wqs_t = oh + (size_t)NB * LL * FF;
  unsigned short* wvs_t = wqs_t + FF * FF;
  unsigned short* fcw_t = wvs_t + FF * FF;
  unsigned short* w1_t  = fcw_t + FF * FF;
  unsigned short* w2_t  = w1_t + DK * DK;                 // [512][64]
  float* out = (float*)d_out;

  pad_zero_kernel<<<dim3(NB * HH), 256, 0, stream>>>(qh, vh_t);
  transpose_bf16_kernel<<<dim3((FF * FF + 255) / 256), 256, 0, stream>>>(w_qs, wqs_t, FF, FF, FF);
  transpose_bf16_kernel<<<dim3((FF * FF + 255) / 256), 256, 0, stream>>>(w_vs, wvs_t, FF, FF, FF);
  transpose_bf16_kernel<<<dim3((FF * FF + 255) / 256), 256, 0, stream>>>(fc_w, fcw_t, FF, FF, FF);
  transpose_bf16_kernel<<<dim3((DK * DK + 255) / 256), 256, 0, stream>>>(w1, w1_t, DK, DK, DK);
  transpose_bf16_kernel<<<dim3((LP * DK + 255) / 256), 256, 0, stream>>>(w2, w2_t, DK, LL, LP);

  proj_kernel<<<dim3((NB * LL) / 64, FF / 64), 128, 0, stream>>>(q, wqs_t, qh, 0);
  proj_kernel<<<dim3((NB * LL) / 64, FF / 64), 128, 0, stream>>>(v, wvs_t, vh_t, 1);
  synth_kernel<<<dim3(LP / 16, NB * HH), 128, 0, stream>>>(qh, vh_t, w1_t, b1, w2_t, b2, oh);
  fc_ln_kernel<<<dim3((NB * LL) / 16), 128, 0, stream>>>(oh, fcw_t, q, ln_g, ln_b, out);
}
